// RWKV6Block_12214886990239
// MI455X (gfx1250) — matrix operand order repaired and checked by reading, not compiled, not run on hardware
//
#include <hip/hip_runtime.h>
#include <math.h>

// ---------------- types ----------------
typedef __attribute__((ext_vector_type(16))) __bf16 v16bf;
typedef __attribute__((ext_vector_type(8)))  __bf16 v8bf;
typedef __attribute__((ext_vector_type(8)))  float  v8f;

#define Bc 4
#define Tc 1024
#define Cc 1024
#define Hc 16
#define Nc 64
#define Fc 3584
#define Mrows (Bc*Tc)          // 4096
#define GN_EPS 64e-5f

__device__ __forceinline__ float sigmf(float x) { return 1.0f / (1.0f + __expf(-x)); }

// ---------------- epilogue ids ----------------
#define EPI_F32        0
#define EPI_TANH_BF16  1
#define EPI_MIX_BF16   2
#define EPI_WDECAY     3
#define EPI_SILU       4
#define EPI_ADDRES     5
#define EPI_SIGMOID    6
#define EPI_RELU2_BF16 7

// ---------------- tiled transpose + f32->bf16: dst[N][K] = src[K][N] ----------------
__global__ void k_convT(const float* __restrict__ src, __bf16* __restrict__ dst,
                        int K, int Nout) {
    __shared__ float tile[32][33];
    const int kb = blockIdx.x * 32, nb = blockIdx.y * 32;
    const int tx = threadIdx.x, ty = threadIdx.y;   // block (32, 8)
#pragma unroll
    for (int yy = ty; yy < 32; yy += 8) {
        int k = kb + yy, n = nb + tx;
        tile[yy][tx] = (k < K && n < Nout) ? src[(size_t)k * Nout + n] : 0.f;
    }
    __syncthreads();
#pragma unroll
    for (int yy = ty; yy < 32; yy += 8) {
        int n = nb + yy, k = kb + tx;
        if (n < Nout && k < K) dst[(size_t)n * K + k] = (__bf16)tile[tx][yy];
    }
}

// ---------------- LayerNorm over C=1024, one row per block (256 thr) ----------------
__global__ void k_layernorm(const float* __restrict__ in,
                            const float* __restrict__ w, const float* __restrict__ b,
                            float* __restrict__ out) {
    __shared__ float s1[256], s2[256];
    int row = blockIdx.x, tid = threadIdx.x;
    const float* p = in + (size_t)row * Cc;
    float v[4], sum = 0.f, sq = 0.f;
#pragma unroll
    for (int j = 0; j < 4; ++j) {
        int c = j * 256 + tid;
        v[j] = p[c];
        sum += v[j]; sq += v[j] * v[j];
    }
    s1[tid] = sum; s2[tid] = sq; __syncthreads();
    for (int st = 128; st > 0; st >>= 1) {
        if (tid < st) { s1[tid] += s1[tid + st]; s2[tid] += s2[tid + st]; }
        __syncthreads();
    }
    float mu  = s1[0] * (1.0f / Cc);
    float var = s2[0] * (1.0f / Cc) - mu * mu;
    float inv = rsqrtf(var + 1e-5f);
    float* q = out + (size_t)row * Cc;
#pragma unroll
    for (int j = 0; j < 4; ++j) {
        int c = j * 256 + tid;
        q[c] = (v[j] - mu) * inv * w[c] + b[c];
    }
}

// ---------------- token shift + x_maa mix -> xx(f32), tmx(bf16) ----------------
__global__ void k_shift_mix(const float* __restrict__ xa, const float* __restrict__ x_maa,
                            float* __restrict__ xx, __bf16* __restrict__ tmx) {
    int idx = blockIdx.x * blockDim.x + threadIdx.x;
    if (idx >= Mrows * Cc) return;
    int row = idx >> 10;
    int t   = row & (Tc - 1);
    int c   = idx & (Cc - 1);
    float cur  = xa[idx];
    float prev = (t > 0) ? xa[idx - Cc] : 0.f;
    float xv   = prev - cur;
    xx[idx]  = xv;
    tmx[idx] = (__bf16)(cur + xv * x_maa[c]);
}

// ---------------- channel-mix shift prep -> kx,rx (bf16) ----------------
__global__ void k_cmix_prep(const float* __restrict__ xc,
                            const float* __restrict__ fk_maa, const float* __restrict__ fr_maa,
                            __bf16* __restrict__ kx, __bf16* __restrict__ rx) {
    int idx = blockIdx.x * blockDim.x + threadIdx.x;
    if (idx >= Mrows * Cc) return;
    int row = idx >> 10;
    int t   = row & (Tc - 1);
    int c   = idx & (Cc - 1);
    float cur  = xc[idx];
    float prev = (t > 0) ? xc[idx - Cc] : 0.f;
    float xv   = prev - cur;
    kx[idx] = (__bf16)(cur + xv * fk_maa[c]);
    rx[idx] = (__bf16)(cur + xv * fr_maa[c]);
}

// ---------------- out = x1 + sigmoid * kv ----------------
__global__ void k_final(const float* __restrict__ x1, const float* __restrict__ sg,
                        const float* __restrict__ kv, float* __restrict__ out) {
    int idx = blockIdx.x * blockDim.x + threadIdx.x;
    if (idx >= Mrows * Cc) return;
    out[idx] = x1[idx] + sg[idx] * kv[idx];
}

// ---------------- WMMA bf16 GEMM ----------------
// C[M,N] = A[M,K](bf16,row,lda) x Bt[N,K](bf16,[N][K])
// Block: 256 thr = 8 waves stacked along M, each wave 32x64 (8 accumulators).
// Shared 64x32 B tile staged in LDS, double-buffered, 1 barrier per K-step.
// Requires M % 256 == 0 for barrier uniformity (all our M are 4096).
template <int EPI>
__global__ void gemm_bf16_wmma(const __bf16* __restrict__ A, int lda,
                               const __bf16* __restrict__ Bt, int ldb,
                               int M, int Nout, int K,
                               float* __restrict__ outF, __bf16* __restrict__ outB, int ldo,
                               const float* __restrict__ e1, const float* __restrict__ e2,
                               const float* __restrict__ ecol) {
    __shared__ __bf16 Bs[2][64 * 32];               // 2 x 4KB ping-pong
    const int wave = threadIdx.x >> 5;
    const int lane = threadIdx.x & 31;
    const int lr = lane & 15;
    const int lh = lane >> 4;
    const int m0 = blockIdx.x * 256 + wave * 32;
    const int n0 = blockIdx.y * 64;

    // cooperative B-tile fill mapping: thread -> (col, 16B chunk)
    const int fcol   = threadIdx.x >> 2;            // 0..63
    const int fchunk = threadIdx.x & 3;             // 0..3 (8 bf16 each)
    int gcol = n0 + fcol; if (gcol >= Nout) gcol = Nout - 1;
    const __bf16* gsrc = Bt + (size_t)gcol * ldb + fchunk * 8;
    __bf16* ldst0 = &Bs[0][fcol * 32 + fchunk * 8];
    __bf16* ldst1 = &Bs[1][fcol * 32 + fchunk * 8];

    const __bf16* Arow0 = A + (size_t)(m0 + lr) * lda;
    const __bf16* Arow1 = A + (size_t)(m0 + 16 + lr) * lda;

    v8f z = {0.f,0.f,0.f,0.f,0.f,0.f,0.f,0.f};
    v8f acc[8];
#pragma unroll
    for (int q = 0; q < 8; ++q) acc[q] = z;

    // preload B tile for k0 = 0
    *(v8bf*)ldst0 = *(const v8bf*)gsrc;
    __syncthreads();

    for (int k0 = 0; k0 < K; k0 += 32) {
        const int p = (k0 >> 5) & 1;
        const bool hasnext = (k0 + 32) < K;
        v8bf nxt;
        if (hasnext) nxt = *(const v8bf*)(gsrc + k0 + 32);

        // A fragments (ISA 16-bit A 16x32 layout), direct from global (L2-resident)
        const __bf16* pa0 = Arow0 + k0 + lh * 8;
        const __bf16* pa1 = Arow1 + k0 + lh * 8;
        __builtin_prefetch(pa0 + 32, 0, 1);          // global_prefetch_b8
        v16bf a0, a1;
        *(v8bf*)&a0       = *(const v8bf*)pa0;
        *((v8bf*)&a0 + 1) = *(const v8bf*)(pa0 + 16);
        *(v8bf*)&a1       = *(const v8bf*)pa1;
        *((v8bf*)&a1 + 1) = *(const v8bf*)(pa1 + 16);

#pragma unroll
        for (int j = 0; j < 4; ++j) {
            v16bf bfrag;
            *(v8bf*)&bfrag       = *(const v8bf*)&Bs[p][(j * 16 + lr) * 32 + lh * 8];
            *((v8bf*)&bfrag + 1) = *(const v8bf*)&Bs[p][(j * 16 + lr) * 32 + lh * 8 + 16];
            acc[j]     = __builtin_amdgcn_wmma_f32_16x16x32_bf16(
                             false, a0, false, bfrag, (short)0, acc[j],     false, false);
            acc[4 + j] = __builtin_amdgcn_wmma_f32_16x16x32_bf16(
                             false, a1, false, bfrag, (short)0, acc[4 + j], false, false);
        }

        if (hasnext) *(v8bf*)(p ? ldst0 : ldst1) = nxt;
        __syncthreads();
    }

    // epilogue (C layout: lane = col, rows = lh*8 + i)
#pragma unroll
    for (int mi = 0; mi < 2; ++mi) {
#pragma unroll
        for (int j = 0; j < 4; ++j) {
            int n = n0 + j * 16 + lr;
            if (n >= Nout) continue;
#pragma unroll
            for (int i = 0; i < 8; ++i) {
                int row = m0 + mi * 16 + lh * 8 + i;
                size_t idx = (size_t)row * ldo + n;
                float v = acc[mi * 4 + j][i];
                if (EPI == EPI_F32)             outF[idx] = v;
                else if (EPI == EPI_TANH_BF16)  outB[idx] = (__bf16)tanhf(v);
                else if (EPI == EPI_MIX_BF16) {
                    size_t fidx = (size_t)row * Cc + n;
                    outB[idx] = (__bf16)(e1[fidx] + e2[fidx] * (ecol[n] + v));
                }
                else if (EPI == EPI_WDECAY)     outF[idx] = __expf(-__expf(ecol[n] + v));
                else if (EPI == EPI_SILU)       outF[idx] = v * sigmf(v);
                else if (EPI == EPI_ADDRES)     outF[idx] = e1[idx] + v;
                else if (EPI == EPI_SIGMOID)    outF[idx] = sigmf(v);
                else if (EPI == EPI_RELU2_BF16) { float t = v > 0.f ? v : 0.f; outB[idx] = (__bf16)(t * t); }
            }
        }
    }
}

// ---------------- WKV6 scan ----------------
// One block per (b,h), 256 threads: thread (g = tid>>6, j = tid&63) owns the
// 16-row slice [g*16, g*16+16) of state column j. Per timestep the (r,k,w,u*k)
// quadruple is staged as one float4 in LDS (one ds_load_b128 per inner step).
__global__ void k_wkv6(const float* __restrict__ r, const float* __restrict__ k,
                       const float* __restrict__ v, const float* __restrict__ w,
                       const float* __restrict__ u, float* __restrict__ y) {
    const int b = blockIdx.x >> 4;     // H = 16
    const int h = blockIdx.x & 15;
    const int tid = threadIdx.x;
    const int j = tid & 63;
    const int g = tid >> 6;            // 0..3
    __shared__ float4 s4[Nc];          // {r, k, w, u*k}
    __shared__ float  vs[Nc];
    __shared__ float  part[4][Nc];

    const float uj = u[h * Nc + j];    // used by g==0 stagers
    float S[16];
#pragma unroll
    for (int ii = 0; ii < 16; ++ii) S[ii] = 0.f;

    size_t idx = ((size_t)b * Tc) * Cc + h * Nc + j;
    for (int t = 0; t < Tc; ++t) {
        if (g == 0) {
            float rj = r[idx], kj = k[idx], wj = w[idx], vj = v[idx];
            s4[j] = make_float4(rj, kj, wj, uj * kj);
            vs[j] = vj;
        }
        __syncthreads();
        const float vjl = vs[j];
        float yp = 0.f;
#pragma unroll
        for (int ii = 0; ii < 16; ++ii) {
            float4 q = s4[g * 16 + ii];
            float kv = q.y * vjl;
            yp += q.x * (S[ii] + q.w * vjl);
            S[ii] = q.z * S[ii] + kv;
        }
        part[g][j] = yp;
        __syncthreads();
        if (g == 0) y[idx] = part[0][j] + part[1][j] + part[2][j] + part[3][j];
        idx += Cc;
    }
}

// ---------------- GroupNorm(H) * lnx + gate, -> bf16 ----------------
__global__ void k_gn_gate(const float* __restrict__ y, const float* __restrict__ gsilu,
                          const float* __restrict__ lnx_w, const float* __restrict__ lnx_b,
                          __bf16* __restrict__ out) {
    __shared__ float hsum[Hc], hsq[Hc];
    int row = blockIdx.x, tid = threadIdx.x;
    if (tid < Hc) { hsum[tid] = 0.f; hsq[tid] = 0.f; }
    __syncthreads();
    const float* p = y + (size_t)row * Cc;
    float v[4];
#pragma unroll
    for (int jj = 0; jj < 4; ++jj) {
        int c = jj * 256 + tid;
        v[jj] = p[c];
        int hh = c >> 6;
        atomicAdd(&hsum[hh], v[jj]);
        atomicAdd(&hsq[hh],  v[jj] * v[jj]);
    }
    __syncthreads();
    __bf16* q = out + (size_t)row * Cc;
    const float* g = gsilu + (size_t)row * Cc;
#pragma unroll
    for (int jj = 0; jj < 4; ++jj) {
        int c = jj * 256 + tid;
        int hh = c >> 6;
        float mu  = hsum[hh] * (1.0f / Nc);
        float var = hsq[hh] * (1.0f / Nc) - mu * mu;
        float nv  = (v[jj] - mu) * rsqrtf(var + GN_EPS) * lnx_w[c] + lnx_b[c];
        q[c] = (__bf16)(nv * g[c]);
    }
}

// ---------------- host helpers ----------------
template <int EPI>
static void launch_gemm(hipStream_t s, const __bf16* A, int lda, const __bf16* Bt, int ldb,
                        int M, int N, int K, float* oF, __bf16* oB, int ldo,
                        const float* e1 = nullptr, const float* e2 = nullptr,
                        const float* ec = nullptr) {
    dim3 g(M / 256, (N + 63) / 64);
    gemm_bf16_wmma<EPI><<<g, 256, 0, s>>>(A, lda, Bt, ldb, M, N, K, oF, oB, ldo, e1, e2, ec);
}

static void launch_convT(hipStream_t s, const float* src, __bf16* dst, int K, int Nout) {
    dim3 g((K + 31) / 32, (Nout + 31) / 32);
    k_convT<<<g, dim3(32, 8), 0, s>>>(src, dst, K, Nout);
}

extern "C" void kernel_launch(void* const* d_in, const int* in_sizes, int n_in,
                              void* d_out, int out_size, void* d_ws, size_t ws_size,
                              hipStream_t stream) {
    const float* x      = (const float*)d_in[0];
    const float* ln0_w  = (const float*)d_in[1];
    const float* ln0_b  = (const float*)d_in[2];
    const float* ln1_w  = (const float*)d_in[3];
    const float* ln1_b  = (const float*)d_in[4];
    const float* ln2_w  = (const float*)d_in[5];
    const float* ln2_b  = (const float*)d_in[6];
    const float* x_maa  = (const float*)d_in[7];
    const float* maa5[5] = { (const float*)d_in[8],  (const float*)d_in[9],
                             (const float*)d_in[10], (const float*)d_in[11],
                             (const float*)d_in[12] };          // w,k,v,r,g
    const float* tm_w1  = (const float*)d_in[13];  // [C,160]
    const float* tm_w2  = (const float*)d_in[14];  // [5,32,C]
    const float* td_w1  = (const float*)d_in[15];  // [C,64]
    const float* td_w2  = (const float*)d_in[16];  // [64,C]
    const float* tdec   = (const float*)d_in[17];  // [C]
    const float* tfaaaa = (const float*)d_in[18];  // [H,N]
    const float* Wr     = (const float*)d_in[19];
    const float* Wk     = (const float*)d_in[20];
    const float* Wv     = (const float*)d_in[21];
    const float* Wg     = (const float*)d_in[22];
    const float* Wo     = (const float*)d_in[23];
    const float* lnx_w  = (const float*)d_in[24];
    const float* lnx_b  = (const float*)d_in[25];
    const float* fk_maa = (const float*)d_in[26];
    const float* fr_maa = (const float*)d_in[27];
    const float* Wk_f   = (const float*)d_in[28];  // [C,F]
    const float* Wv_f   = (const float*)d_in[29];  // [F,C]
    const float* Wr_f   = (const float*)d_in[30];

    char* ws = (char*)d_ws;
    size_t off = 0;
    auto alloc = [&](size_t bytes) -> void* {
        off = (off + 255) & ~(size_t)255;
        void* p = ws + off;
        off += bytes;
        return p;
    };
    const size_t MC = (size_t)Mrows * Cc;

    // bf16 transposed weights [N][K]
    __bf16* wt_r   = (__bf16*)alloc((size_t)Cc * Cc * 2);
    __bf16* wt_k   = (__bf16*)alloc((size_t)Cc * Cc * 2);
    __bf16* wt_v   = (__bf16*)alloc((size_t)Cc * Cc * 2);
    __bf16* wt_g   = (__bf16*)alloc((size_t)Cc * Cc * 2);
    __bf16* wt_o   = (__bf16*)alloc((size_t)Cc * Cc * 2);
    __bf16* wt_tm1 = (__bf16*)alloc((size_t)160 * Cc * 2);
    __bf16* wt_tm2[5];
    for (int f = 0; f < 5; ++f) wt_tm2[f] = (__bf16*)alloc((size_t)Cc * 32 * 2);
    __bf16* wt_td1 = (__bf16*)alloc((size_t)64 * Cc * 2);
    __bf16* wt_td2 = (__bf16*)alloc((size_t)Cc * 64 * 2);
    __bf16* wt_kf  = (__bf16*)alloc((size_t)Fc * Cc * 2);
    __bf16* wt_vf  = (__bf16*)alloc((size_t)Cc * Fc * 2);
    __bf16* wt_rf  = (__bf16*)alloc((size_t)Cc * Cc * 2);

    // activations
    float*  x0    = (float*)alloc(MC * 4);
    float*  xa    = (float*)alloc(MC * 4);
    float*  xx    = (float*)alloc(MC * 4);
    __bf16* tmx   = (__bf16*)alloc(MC * 2);
    __bf16* tanh5 = (__bf16*)alloc((size_t)Mrows * 160 * 2);
    __bf16* mixb[5];
    for (int f = 0; f < 5; ++f) mixb[f] = (__bf16*)alloc(MC * 2);  // xw,xk,xv,xr,xg
    float*  rbuf  = (float*)alloc(MC * 4);
    float*  kbuf  = (float*)alloc(MC * 4);
    float*  vbuf  = (float*)alloc(MC * 4);
    float*  gbuf  = (float*)alloc(MC * 4);
    __bf16* wtanh = (__bf16*)alloc((size_t)Mrows * 64 * 2);
    float*  wbuf  = (float*)alloc(MC * 4);
    float*  ybuf  = (float*)alloc(MC * 4);
    __bf16* ygate = (__bf16*)alloc(MC * 2);
    float*  x1    = (float*)alloc(MC * 4);
    float*  xc    = (float*)alloc(MC * 4);
    __bf16* kxb   = (__bf16*)alloc(MC * 2);
    __bf16* rxb   = (__bf16*)alloc(MC * 2);
    __bf16* kkb   = (__bf16*)alloc((size_t)Mrows * Fc * 2);
    float*  kvbuf = (float*)alloc(MC * 4);
    float*  sgbuf = (float*)alloc(MC * 4);

    // ---- weight transposes (f32 -> bf16, [N][K]) ----
    launch_convT(stream, Wr, wt_r, Cc, Cc);
    launch_convT(stream, Wk, wt_k, Cc, Cc);
    launch_convT(stream, Wv, wt_v, Cc, Cc);
    launch_convT(stream, Wg, wt_g, Cc, Cc);
    launch_convT(stream, Wo, wt_o, Cc, Cc);
    launch_convT(stream, tm_w1, wt_tm1, Cc, 160);
    for (int f = 0; f < 5; ++f)
        launch_convT(stream, tm_w2 + (size_t)f * 32 * Cc, wt_tm2[f], 32, Cc);
    launch_convT(stream, td_w1, wt_td1, Cc, 64);
    launch_convT(stream, td_w2, wt_td2, 64, Cc);
    launch_convT(stream, Wk_f, wt_kf, Cc, Fc);
    launch_convT(stream, Wv_f, wt_vf, Fc, Cc);
    launch_convT(stream, Wr_f, wt_rf, Cc, Cc);

    const int EW_BLOCKS = (int)((MC + 255) / 256);

    // ---- ln0, ln1, token shift ----
    k_layernorm<<<Mrows, 256, 0, stream>>>(x,  ln0_w, ln0_b, x0);
    k_layernorm<<<Mrows, 256, 0, stream>>>(x0, ln1_w, ln1_b, xa);
    k_shift_mix<<<EW_BLOCKS, 256, 0, stream>>>(xa, x_maa, xx, tmx);

    // ---- token-shift LoRA ----
    launch_gemm<EPI_TANH_BF16>(stream, tmx, Cc, wt_tm1, Cc, Mrows, 160, Cc,
                               nullptr, tanh5, 160);
    for (int f = 0; f < 5; ++f)
        launch_gemm<EPI_MIX_BF16>(stream, tanh5 + f * 32, 160, wt_tm2[f], 32,
                                  Mrows, Cc, 32, nullptr, mixb[f], Cc,
                                  xa, xx, maa5[f]);

    // ---- projections ----
    launch_gemm<EPI_F32 >(stream, mixb[3], Cc, wt_r, Cc, Mrows, Cc, Cc, rbuf, nullptr, Cc);
    launch_gemm<EPI_F32 >(stream, mixb[1], Cc, wt_k, Cc, Mrows, Cc, Cc, kbuf, nullptr, Cc);
    launch_gemm<EPI_F32 >(stream, mixb[2], Cc, wt_v, Cc, Mrows, Cc, Cc, vbuf, nullptr, Cc);
    launch_gemm<EPI_SILU>(stream, mixb[4], Cc, wt_g, Cc, Mrows, Cc, Cc, gbuf, nullptr, Cc);

    // ---- decay LoRA: w = exp(-exp(time_decay + tanh(xw@td_w1)@td_w2)) ----
    launch_gemm<EPI_TANH_BF16>(stream, mixb[0], Cc, wt_td1, Cc, Mrows, 64, Cc,
                               nullptr, wtanh, 64);
    launch_gemm<EPI_WDECAY>(stream, wtanh, 64, wt_td2, 64, Mrows, Cc, 64,
                            wbuf, nullptr, Cc, nullptr, nullptr, tdec);

    // ---- WKV6 scan ----
    k_wkv6<<<Bc * Hc, 256, 0, stream>>>(rbuf, kbuf, vbuf, wbuf, tfaaaa, ybuf);

    // ---- GroupNorm + gate, output proj + residual ----
    k_gn_gate<<<Mrows, 256, 0, stream>>>(ybuf, gbuf, lnx_w, lnx_b, ygate);
    launch_gemm<EPI_ADDRES>(stream, ygate, Cc, wt_o, Cc, Mrows, Cc, Cc,
                            x1, nullptr, Cc, x0);

    // ---- ChannelMix ----
    k_layernorm<<<Mrows, 256, 0, stream>>>(x1, ln2_w, ln2_b, xc);
    k_cmix_prep<<<EW_BLOCKS, 256, 0, stream>>>(xc, fk_maa, fr_maa, kxb, rxb);
    launch_gemm<EPI_RELU2_BF16>(stream, kxb, Cc, wt_kf, Cc, Mrows, Fc, Cc,
                                nullptr, kkb, Fc);
    launch_gemm<EPI_F32>(stream, kkb, Fc, wt_vf, Fc, Mrows, Cc, Fc, kvbuf, nullptr, Cc);
    launch_gemm<EPI_SIGMOID>(stream, rxb, Cc, wt_rf, Cc, Mrows, Cc, Cc, sgbuf, nullptr, Cc);
    k_final<<<EW_BLOCKS, 256, 0, stream>>>(x1, sgbuf, kvbuf, (float*)d_out);

    (void)in_sizes; (void)n_in; (void)out_size; (void)ws_size;
}
